// Gpt2Attention_62972810494379
// MI455X (gfx1250) — compile-verified, matrix operand order checked by reading, not run on hardware
//
#include <hip/hip_runtime.h>
#include <hip/hip_bf16.h>

// ---------------- problem dims (GPT-2 small attention) ----------------
constexpr int Bz = 4, Sz = 2048, Ez = 768, Hz = 12, Dz = 64;
constexpr int Mz = Bz * Sz;       // 8192
constexpr int NQKV = 3 * Ez;      // 2304

// ---------------- feature detection (gfx1250 async / TDM paths) ----------------
#if defined(__has_builtin)
#  if __has_builtin(__builtin_amdgcn_global_load_async_to_lds_b128)
#    define HAVE_ASYNC 1
#  endif
#  if __has_builtin(__builtin_amdgcn_tensor_load_to_lds)
#    define HAVE_TDM 1
#  endif
#endif

#if defined(__has_builtin) && __has_builtin(__builtin_amdgcn_s_wait_asynccnt)
#  define WAIT_ASYNC() __builtin_amdgcn_s_wait_asynccnt(0)
#else
#  define WAIT_ASYNC() asm volatile("s_wait_asynccnt 0" ::: "memory")
#endif

#if defined(__has_builtin) && __has_builtin(__builtin_amdgcn_s_wait_tensorcnt)
#  define WAIT_TENSOR() __builtin_amdgcn_s_wait_tensorcnt(0)
#else
#  define WAIT_TENSOR() asm volatile("s_wait_tensorcnt 0" ::: "memory")
#endif

// The async-to-LDS builtins take generic pointers to int4 vectors.
typedef int v4i_t __attribute__((ext_vector_type(4)));

// LDS byte offset of a generic pointer to a __shared__ object:
// per the aperture rules addr[31:0] is the LDS offset.
__device__ __forceinline__ unsigned int lds_off(const void* p) {
  return (unsigned int)(size_t)p;
}

#if defined(HAVE_TDM)
typedef unsigned int u32x4 __attribute__((ext_vector_type(4)));
typedef int          i32x4 __attribute__((ext_vector_type(4)));
typedef int          i32x8 __attribute__((ext_vector_type(8)));

__device__ __forceinline__ void tdm_load_2d(u32x4 g0, i32x8 g1) {
  i32x4 z4 = {0, 0, 0, 0};
#if __clang_major__ >= 23
  i32x8 z8 = {0, 0, 0, 0, 0, 0, 0, 0};
  __builtin_amdgcn_tensor_load_to_lds(g0, g1, z4, z4, z8, 0);
#else
  __builtin_amdgcn_tensor_load_to_lds(g0, g1, z4, z4, 0);
#endif
}
#endif

// ---------------- low-level types ----------------
typedef __attribute__((ext_vector_type(16))) __bf16 v16bf;
typedef __attribute__((ext_vector_type(8)))  float  v8f;

struct alignas(16) U128 { unsigned int w[4]; };
struct alignas(16) F4   { float f[4]; };

union AFrag { v16bf v; U128 u[2]; };
union U8x16 { U128 u; unsigned short s[8]; };

__device__ __forceinline__ unsigned short f2bf(float f) {
  unsigned int x = __float_as_uint(f);
  unsigned int r = (x + 0x7FFFu + ((x >> 16) & 1u)) >> 16;  // RNE
  return (unsigned short)r;
}

__device__ __forceinline__ v8f wmma_bf16(const AFrag& a, const AFrag& b, v8f c) {
  return __builtin_amdgcn_wmma_f32_16x16x32_bf16(
      /*neg_a=*/false, a.v, /*neg_b=*/false, b.v,
      /*c_mod=*/(short)0, c, /*reuse_a=*/false, /*reuse_b=*/false);
}

// ---------------- f32 -> bf16 convert (8 elems / thread) ----------------
__global__ void cvt_f32_bf16_k(const float* __restrict__ in,
                               unsigned short* __restrict__ out, int n8) {
  int i = blockIdx.x * blockDim.x + threadIdx.x;
  if (i >= n8) return;
  F4 a = ((const F4*)in)[2 * i];
  F4 b = ((const F4*)in)[2 * i + 1];
  U8x16 o;
#pragma unroll
  for (int c = 0; c < 4; ++c) { o.s[c] = f2bf(a.f[c]); o.s[4 + c] = f2bf(b.f[c]); }
  ((U128*)out)[i] = o.u;
}

// ---------------- bf16 GEMM: C = A(MxK) * B(KxN) + bias ----------------
// Block tile 128x128, BK=32. 8 waves, each wave -> 32x64 (2x4 WMMA accs).
// A-tile staged by the Tensor Data Mover (one 2D descriptor per K-step, with
// LDS padding injected by TDM to get the 80B row stride), falling back to
// per-lane async-to-LDS, then to a manual copy.
// EPI==0: scatter bf16 into Q/K/V [B,H,S,D].  EPI==1: f32 to Of row-major.
template <int EPI>
__global__ __launch_bounds__(256) void gemm_bf16_k(
    const unsigned short* __restrict__ A, const unsigned short* __restrict__ Bm,
    const float* __restrict__ bias,
    unsigned short* __restrict__ Oq, unsigned short* __restrict__ Ok,
    unsigned short* __restrict__ Ov, float* __restrict__ Of,
    int M, int N, int K) {
  const int tid  = threadIdx.x;
  const int lane = tid & 31;
  const int w    = tid >> 5;
  const int wm   = (w & 3) * 32;
  const int wn   = (w >> 2) * 64;
  const int bm   = blockIdx.y * 128;
  const int bn   = blockIdx.x * 128;
  const int lm   = lane & 15;
  const int hi   = lane >> 4;
  const int koff = hi * 8;

  __shared__ unsigned short As[128][40];  // row-major [m][k], stride 80B (16B-aligned)
  __shared__ unsigned short Bt[128][40];  // transposed [n][k]

  v8f acc[2][4];
#pragma unroll
  for (int mi = 0; mi < 2; ++mi)
#pragma unroll
    for (int ni = 0; ni < 4; ++ni)
#pragma unroll
      for (int j = 0; j < 8; ++j) acc[mi][ni][j] = 0.0f;

  // staging decomposition
  const int aM = tid >> 2;           // 0..63
  const int aK = (tid & 3) * 8;      // 0,8,16,24
  const int bN = (tid & 15) * 8;     // 0..120
  const int bK = tid >> 4;           // 0..15

  for (int k0 = 0; k0 < K; k0 += 32) {
    __syncthreads();
    // ---- stage A (128x32 bf16 tile) ----
#if defined(HAVE_TDM)
    if (tid < 32) {  // one TDM descriptor per K-step, issued by wave 0
      unsigned long long ga =
          (unsigned long long)(size_t)&A[(size_t)bm * K + k0];
      u32x4 g0;
      g0[0] = 1u;                                  // count=1, user mode
      g0[1] = lds_off(&As[0][0]);                  // lds_addr
      g0[2] = (unsigned int)(ga & 0xFFFFFFFFu);    // global_addr[31:0]
      g0[3] = (unsigned int)((ga >> 32) & 0x1FFFFFFu) | (2u << 30);  // addr hi | type=2
      i32x8 g1;
      // data_size=2B (code 1), pad_enable=1, pad_interval=16 DW (code 3),
      // pad_amount=4 DW (code 3)  -> row stride 64B + 16B = 80B in LDS
      g1[0] = (1 << 16) | (1 << 20) | (3 << 22) | (3 << 25);
      g1[1] = (32  & 0xFFFF) << 16;                // tensor_dim0 = 32
      g1[2] = (128 & 0xFFFF) << 16;                // tensor_dim1 = 128
      g1[3] = 32 << 16;                            // tile_dim0 = 32
      g1[4] = 128;                                 // tile_dim1 = 128 (tile_dim2 = 0)
      g1[5] = K;                                   // tensor_dim0_stride = K
      g1[6] = 0;
      g1[7] = 0;
      tdm_load_2d(g0, g1);
      WAIT_TENSOR();
    }
#elif defined(HAVE_ASYNC)
#pragma unroll
    for (int i = 0; i < 2; ++i) {
      int row = aM + i * 64;
      __builtin_amdgcn_global_load_async_to_lds_b128(
          (v4i_t*)&A[(size_t)(bm + row) * K + k0 + aK],
          (v4i_t*)&As[row][aK], 0, 0);
    }
#else
#pragma unroll
    for (int i = 0; i < 2; ++i) {
      int row = aM + i * 64;
      *(U128*)&As[row][aK] =
          *(const U128*)&A[(size_t)(bm + row) * K + k0 + aK];
    }
#endif
    // ---- stage B transposed (32x128 -> [n][k]) ----
#pragma unroll
    for (int i = 0; i < 2; ++i) {
      int k = bK + i * 16;
      U8x16 t;
      t.u = *(const U128*)&Bm[(size_t)(k0 + k) * N + bn + bN];
#pragma unroll
      for (int c = 0; c < 8; ++c) Bt[bN + c][k] = t.s[c];
    }
    if (k0 + 32 < K) {
      __builtin_prefetch(&A[(size_t)(bm + aM) * K + k0 + 32], 0, 1);
      __builtin_prefetch(&Bm[(size_t)(k0 + 32 + bK) * N + bn + bN], 0, 1);
    }
#if !defined(HAVE_TDM) && defined(HAVE_ASYNC)
    WAIT_ASYNC();
#endif
    __syncthreads();

    // ---- compute ----
    AFrag af[2], bf[4];
#pragma unroll
    for (int mi = 0; mi < 2; ++mi) {
      int row = wm + mi * 16 + lm;
      af[mi].u[0] = *(const U128*)&As[row][koff];
      af[mi].u[1] = *(const U128*)&As[row][koff + 16];
    }
#pragma unroll
    for (int ni = 0; ni < 4; ++ni) {
      int col = wn + ni * 16 + lm;
      bf[ni].u[0] = *(const U128*)&Bt[col][koff];
      bf[ni].u[1] = *(const U128*)&Bt[col][koff + 16];
    }
#pragma unroll
    for (int mi = 0; mi < 2; ++mi)
#pragma unroll
      for (int ni = 0; ni < 4; ++ni)
        acc[mi][ni] = wmma_bf16(af[mi], bf[ni], acc[mi][ni]);
  }

  // ---- epilogue ----
#pragma unroll
  for (int mi = 0; mi < 2; ++mi) {
#pragma unroll
    for (int ni = 0; ni < 4; ++ni) {
      int colg = bn + wn + ni * 16 + lm;
      float bv = bias[colg];
#pragma unroll
      for (int j = 0; j < 8; ++j) {
        int rowg = bm + wm + mi * 16 + hi * 8 + j;
        float v = acc[mi][ni][j] + bv;
        if (EPI == 0) {
          int which = colg / Ez;
          int e = colg - which * Ez;
          int h = e >> 6, d = e & 63;
          int b = rowg >> 11, s = rowg & 2047;
          unsigned short* dst = (which == 0) ? Oq : (which == 1) ? Ok : Ov;
          dst[(((size_t)b * Hz + h) * Sz + s) * Dz + d] = f2bf(v);
        } else {
          Of[(size_t)rowg * N + colg] = v;
        }
      }
    }
  }
}

// ---------------- flash attention (causal), bf16 QKV in [B,H,S,D] ----------------
// Block: 256 threads (8 waves) handles 128 q rows of one (b,h).
// K tile staged via async global->LDS (ASYNCcnt), V transposed through VGPRs.
__global__ __launch_bounds__(256) void flash_attn_k(
    const unsigned short* __restrict__ Q, const unsigned short* __restrict__ Kg,
    const unsigned short* __restrict__ Vg, unsigned short* __restrict__ O) {
  const int qb = blockIdx.x * 128;
  const int h  = blockIdx.y;
  const int b  = blockIdx.z;
  const int tid = threadIdx.x, lane = tid & 31, w = tid >> 5;
  const int lm = lane & 15, hi = lane >> 4, koff = hi * 8;
  const size_t bh = ((size_t)b * Hz + h) * Sz;

  __shared__ unsigned short sK[32][64];       // [key][d]
  __shared__ unsigned short sVt[64][32];      // [d][key]
  __shared__ unsigned short sP[8][16][32];    // per-wave P tile

  // Q fragments (d 0..31 and 32..63), registers for whole kernel
  AFrag qf[2];
  {
    const unsigned short* qp = Q + (bh + qb + w * 16 + lm) * Dz;
    qf[0].u[0] = *(const U128*)(qp + koff);
    qf[0].u[1] = *(const U128*)(qp + koff + 16);
    qf[1].u[0] = *(const U128*)(qp + 32 + koff);
    qf[1].u[1] = *(const U128*)(qp + 32 + koff + 16);
  }

  v8f o[4];
  float m[8], l[8];
#pragma unroll
  for (int j = 0; j < 8; ++j) { m[j] = -1e30f; l[j] = 0.0f; }
#pragma unroll
  for (int t = 0; t < 4; ++t)
#pragma unroll
    for (int j = 0; j < 8; ++j) o[t][j] = 0.0f;

  const float scale = 0.125f;  // 1/sqrt(64)
  const int kend = qb + 128;   // causal limit for this block

  for (int kb = 0; kb < kend; kb += 32) {
    __syncthreads();
    // ---- cooperative stage of K tile (async) and transposed V tile ----
    {
      int r  = tid >> 3;          // key row 0..31
      int dO = (tid & 7) * 8;     // d offset 0..56
      const unsigned short* kp = Kg + (bh + kb + r) * Dz + dO;
      const unsigned short* vp = Vg + (bh + kb + r) * Dz + dO;
#if defined(HAVE_ASYNC)
      __builtin_amdgcn_global_load_async_to_lds_b128(
          (v4i_t*)kp, (v4i_t*)&sK[r][dO], 0, 0);
#else
      *(U128*)&sK[r][dO] = *(const U128*)kp;
#endif
      U8x16 vv; vv.u = *(const U128*)vp;
#pragma unroll
      for (int c = 0; c < 8; ++c) sVt[dO + c][r] = vv.s[c];
    }
#if defined(HAVE_ASYNC)
    WAIT_ASYNC();
#endif
    __syncthreads();

    // ---- scores: two 16x16 tiles over 32 keys ----
    v8f c0, c1;
#pragma unroll
    for (int j = 0; j < 8; ++j) { c0[j] = 0.0f; c1[j] = 0.0f; }
    AFrag kf;
#pragma unroll
    for (int c = 0; c < 2; ++c) {
      kf.u[0] = *(const U128*)&sK[lm][c * 32 + koff];
      kf.u[1] = *(const U128*)&sK[lm][c * 32 + koff + 16];
      c0 = wmma_bf16(qf[c], kf, c0);
    }
#pragma unroll
    for (int c = 0; c < 2; ++c) {
      kf.u[0] = *(const U128*)&sK[16 + lm][c * 32 + koff];
      kf.u[1] = *(const U128*)&sK[16 + lm][c * 32 + koff + 16];
      c1 = wmma_bf16(qf[c], kf, c1);
    }

    // ---- causal mask + online softmax (rows live in VGPR j across 16-lane halves) ----
    const int rbase = qb + w * 16 + hi * 8;
#pragma unroll
    for (int j = 0; j < 8; ++j) {
      int row  = rbase + j;
      float s0 = (kb + lm      <= row) ? c0[j] * scale : -1e30f;
      float s1 = (kb + 16 + lm <= row) ? c1[j] * scale : -1e30f;
      float mx = fmaxf(s0, s1);
#pragma unroll
      for (int off = 8; off; off >>= 1) mx = fmaxf(mx, __shfl_xor(mx, off, 32));
      float mn    = fmaxf(m[j], mx);
      float alpha = __expf(m[j] - mn);
      float p0    = __expf(s0 - mn);
      float p1    = __expf(s1 - mn);
      float rs    = p0 + p1;
#pragma unroll
      for (int off = 8; off; off >>= 1) rs += __shfl_xor(rs, off, 32);
      l[j] = l[j] * alpha + rs;
      m[j] = mn;
#pragma unroll
      for (int t = 0; t < 4; ++t) o[t][j] *= alpha;
      sP[w][hi * 8 + j][lm]      = f2bf(p0);
      sP[w][hi * 8 + j][16 + lm] = f2bf(p1);
    }

    // wave-private LDS round trip: C-layout -> A-layout for P
    asm volatile("s_wait_dscnt 0" ::: "memory");
    AFrag pf;
    pf.u[0] = *(const U128*)&sP[w][lm][koff];
    pf.u[1] = *(const U128*)&sP[w][lm][koff + 16];

    // ---- O += P(16x32) @ V(32x64) ----
    AFrag vf;
#pragma unroll
    for (int t = 0; t < 4; ++t) {
      vf.u[0] = *(const U128*)&sVt[t * 16 + lm][koff];
      vf.u[1] = *(const U128*)&sVt[t * 16 + lm][koff + 16];
      o[t] = wmma_bf16(pf, vf, o[t]);
    }
  }

  // ---- normalize and store attn out (bf16, [B,S,E]) ----
  float inv[8];
#pragma unroll
  for (int j = 0; j < 8; ++j) inv[j] = 1.0f / l[j];
#pragma unroll
  for (int t = 0; t < 4; ++t) {
#pragma unroll
    for (int j = 0; j < 8; ++j) {
      int row = qb + w * 16 + hi * 8 + j;
      int col = h * Dz + t * 16 + lm;
      O[((size_t)b * Sz + row) * Ez + col] = f2bf(o[t][j] * inv[j]);
    }
  }
}

// ---------------- host-side launch ----------------
extern "C" void kernel_launch(void* const* d_in, const int* in_sizes, int n_in,
                              void* d_out, int out_size, void* d_ws, size_t ws_size,
                              hipStream_t stream) {
  (void)in_sizes; (void)n_in; (void)out_size; (void)ws_size;
  const float* x      = (const float*)d_in[0];
  const float* w_qkv  = (const float*)d_in[1];
  const float* b_qkv  = (const float*)d_in[2];
  const float* w_proj = (const float*)d_in[3];
  const float* b_proj = (const float*)d_in[4];
  float* out = (float*)d_out;

  char* ws = (char*)d_ws;
  size_t off = 0;
  auto alloc = [&](size_t bytes) {
    void* p = ws + off;
    off += (bytes + 255) & ~(size_t)255;
    return p;
  };
  const size_t nX  = (size_t)Mz * Ez;        // 6291456
  const size_t nWq = (size_t)Ez * NQKV;      // 1769472
  const size_t nWp = (size_t)Ez * Ez;        // 589824

  unsigned short* xb    = (unsigned short*)alloc(nX * 2);
  unsigned short* wqkvb = (unsigned short*)alloc(nWq * 2);
  unsigned short* wprjb = (unsigned short*)alloc(nWp * 2);
  unsigned short* Qb    = (unsigned short*)alloc(nX * 2);
  unsigned short* Kb    = (unsigned short*)alloc(nX * 2);
  unsigned short* Vb    = (unsigned short*)alloc(nX * 2);
  unsigned short* attnb = (unsigned short*)alloc(nX * 2);

  // 1) convert inputs to bf16
  {
    int n8 = (int)(nX / 8);
    cvt_f32_bf16_k<<<(n8 + 255) / 256, 256, 0, stream>>>(x, xb, n8);
    n8 = (int)(nWq / 8);
    cvt_f32_bf16_k<<<(n8 + 255) / 256, 256, 0, stream>>>(w_qkv, wqkvb, n8);
    n8 = (int)(nWp / 8);
    cvt_f32_bf16_k<<<(n8 + 255) / 256, 256, 0, stream>>>(w_proj, wprjb, n8);
  }

  // 2) QKV projection: (8192x768)x(768x2304) -> Q/K/V bf16 [B,H,S,D]
  gemm_bf16_k<0><<<dim3(NQKV / 128, Mz / 128), 256, 0, stream>>>(
      xb, wqkvb, b_qkv, Qb, Kb, Vb, nullptr, Mz, NQKV, Ez);

  // 3) causal flash attention -> attn bf16 [B,S,E]
  flash_attn_k<<<dim3(Sz / 128, Hz, Bz), 256, 0, stream>>>(Qb, Kb, Vb, attnb);

  // 4) output projection: (8192x768)x(768x768) + bias -> f32 out
  gemm_bf16_k<1><<<dim3(Ez / 128, Mz / 128), 256, 0, stream>>>(
      attnb, wprjb, b_proj, nullptr, nullptr, nullptr, out, Mz, Ez, Ez);
}
